// GCN_TRAJ_13855564497014
// MI455X (gfx1250) — matrix operand order repaired and checked by reading, not compiled, not run on hardware
//
#include <hip/hip_runtime.h>
#include <math.h>

typedef __attribute__((ext_vector_type(16))) _Float16 v16h;
typedef __attribute__((ext_vector_type(8)))  _Float16 v8h;
typedef __attribute__((ext_vector_type(4)))  _Float16 v4h;
typedef __attribute__((ext_vector_type(8)))  float    v8f;
typedef __attribute__((ext_vector_type(4)))  float    v4f;

union ABu { v16h v; v8h h[2]; };

#define NHD   256
#define NV    21
#define NTC   60
#define NCC   4
#define NDCT  60
#define NTHIS 30
#define NBS   256
#define NTV   (NTC*NV)      // 1260
#define NPAD  1280          // 80*16 -> exactly 10 n-tiles per wave (even pairing)
#define NTILES_N 80
#define NOV   336           // 16 o * 21 v rows of sHT (21 M-tiles exactly)

// fast tanh: 1 - 2/(exp2(2*log2e*x)+1); v_exp_f32 + v_rcp_f32, saturates at +/-1
__device__ __forceinline__ float fast_tanh(float x) {
  float e = __builtin_amdgcn_exp2f(x * 2.885390081777927f);
  return 1.f - 2.f * __builtin_amdgcn_rcpf(e + 1.f);
}

// ---------------------------------------------------------------------------
// Kernel 1: batched DCT of padded root history.
// ---------------------------------------------------------------------------
__global__ void __launch_bounds__(192) k_dct(const float* __restrict__ x,
                                             const float* __restrict__ dct_m,
                                             const int* __restrict__ root_idx,
                                             float* __restrict__ ws_dct) {
  int b = blockIdx.x;
  int tid = threadIdx.x;
  if (tid >= NDCT * 3) return;
  int d = tid / 3, k = tid % 3;
  int ri = root_idx[k];
  float last = x[29 * NBS * 63 + b * 63 + ri];
  float acc = 0.f, tail = 0.f;
  for (int t = 0; t < NTHIS; ++t)
    acc += dct_m[d * 90 + t] * x[t * NBS * 63 + b * 63 + ri];
  for (int t = NTHIS; t < 90; ++t)
    tail += dct_m[d * 90 + t];
  ws_dct[b * 180 + tid] = acc + tail * last;
}

// ---------------------------------------------------------------------------
// Kernel 2: pack head-layer weights into WMMA-B-fragment-ready f16 tiles.
// ---------------------------------------------------------------------------
__global__ void __launch_bounds__(256) k_pack(const float* __restrict__ W,
                                              _Float16* __restrict__ dst,
                                              int in_dim, int out_dim) {
  int ktiles = (in_dim + 31) / 32;
  int ntiles = (out_dim + 15) / 16;
  int total = ntiles * ktiles * 512;
  for (int i = blockIdx.x * 256 + threadIdx.x; i < total; i += gridDim.x * 256) {
    int tile = i >> 9, rem = i & 511;
    int lane = rem >> 4, h = rem & 15;
    int nt = tile / ktiles, kt = tile % ktiles;
    int o = nt * 16 + (lane & 15);
    int k = kt * 32 + ((lane >> 4) << 3) + (h & 7) + ((h >> 3) << 4);
    float wv = (o < out_dim && k < in_dim) ? W[o * in_dim + k] : 0.f;
    dst[i] = (_Float16)wv;
  }
}

// ---------------------------------------------------------------------------
// Kernel 3: fused STSGCN + time encoder + node encoder.  One block per b.
// Main GEMM (K=8) writes PReLU'd h in A-fragment-ready f16 layout sHT[ov][64];
// te1 is then 21x2 WMMAs per o-tile with te1^T B-fragments held in registers.
// ---------------------------------------------------------------------------
struct PreBufs { float xc[NCC][NTV]; float g1[NCC][NTV]; };      // 40,320 B
union GcnOvl { PreBufs pre; _Float16 ht[NOV][64]; };             // 43,008 B

__global__ void __launch_bounds__(256)
k_gcn(const float* __restrict__ cont,
      const float* __restrict__ T_c, const float* __restrict__ A_c,
      const float* __restrict__ conv_w, const float* __restrict__ conv_b,
      const float* __restrict__ bn1_scale, const float* __restrict__ bn1_shift,
      const float* __restrict__ resconv_w, const float* __restrict__ resconv_b,
      const float* __restrict__ bnr_scale, const float* __restrict__ bnr_shift,
      const float* __restrict__ prelu_a_p,
      const float* __restrict__ te1_w, const float* __restrict__ te1_b,
      const float* __restrict__ te2_w, const float* __restrict__ te2_b,
      const float* __restrict__ ne1_w, const float* __restrict__ ne1_b,
      const float* __restrict__ ne2_w, const float* __restrict__ ne2_b,
      float* __restrict__ ws_hcont) {
  __shared__ GcnOvl   sOv;                 // {xc,g1} then ht[336][64] f16
  __shared__ _Float16 sU[NPAD][8];         // B operand, column-major: 16B per column
  __shared__ _Float16 sW8[NHD][8];         // BN-folded conv weights, 16B per o
  __shared__ float    sCb[NHD];            // folded bias
  __shared__ float    sZ1[NOV][10];        // tanh(te1) per (ov, j)
  __shared__ float    sZ2[16][NV];
  __shared__ float    sNe1w[10 * NV];
  __shared__ float    sTeB[10], sTe2w[10], sNeB[10], sNe2w[10];
  __shared__ float    sTe2b, sNe2b;
  __shared__ _Float16 sZeroH[8];           // 16B of zeros (fragment stub)

  const int b   = blockIdx.x;
  const int tid = threadIdx.x;
  const int lane = tid & 31;
  const int wave = tid >> 5;
  const int lh = lane & 15;
  const int hi = (lane >> 4) & 1;
  const float pa = prelu_a_p[0];

  for (int i = tid; i < 10 * NV; i += 256) sNe1w[i] = ne1_w[i];
  if (tid < 10) {
    sTeB[tid]  = te1_b[tid]; sTe2w[tid] = te2_w[tid];
    sNeB[tid]  = ne1_b[tid]; sNe2w[tid] = ne2_w[tid];
  }
  if (tid < 8) sZeroH[tid] = (_Float16)0.f;
  if (tid == 0) { sTe2b = te2_b[0]; sNe2b = ne2_b[0]; }

  // BN-folded 1x1 conv weights + bias
  {
    int o = tid;
    float s1 = bn1_scale[o], sr = bnr_scale[o];
    v8h w;
#pragma unroll
    for (int c = 0; c < 4; ++c) {
      w[c]     = (_Float16)(s1 * conv_w[o * 4 + c]);
      w[4 + c] = (_Float16)(sr * resconv_w[o * 4 + c]);
    }
    *(v8h*)&sW8[o][0] = w;
    sCb[o] = s1 * conv_b[o] + bn1_shift[o] + sr * resconv_b[o] + bnr_shift[o];
  }

  // xc[c,t,v] = cont[b, t, v*4+c]
  for (int i = tid; i < NTV * NCC; i += 256) {
    int c = i & 3, tv = i >> 2;
    sOv.pre.xc[c][tv] = cont[b * NTV * NCC + i];
  }
  __syncthreads();

  // g1[c,q,v] = sum_t xc[c,t,v] * T_c[v,t,q]   (T_c load shared across c)
  for (int p = tid; p < NTV; p += 256) {
    int q = p / NV, v = p % NV;
    float acc[4] = {0.f, 0.f, 0.f, 0.f};
    for (int t = 0; t < NTC; ++t) {
      float tw = T_c[(v * NTC + t) * NTC + q];
#pragma unroll
      for (int c = 0; c < 4; ++c) acc[c] += sOv.pre.xc[c][t * NV + v] * tw;
    }
#pragma unroll
    for (int c = 0; c < 4; ++c) sOv.pre.g1[c][q * NV + v] = acc[c];
  }
  __syncthreads();

  // u columns: k<4 -> sum_v g1[k,t,v]*A_c[t,v,w] ; k>=4 -> xc.  16B store per col.
  for (int n = tid; n < NPAD; n += 256) {
    v8h col = {};
    if (n < NTV) {
      int t = n / NV, w = n % NV;
      float acc[4] = {0.f, 0.f, 0.f, 0.f};
      for (int v = 0; v < NV; ++v) {
        float aw = A_c[(t * NV + v) * NV + w];
#pragma unroll
        for (int k = 0; k < 4; ++k) acc[k] += sOv.pre.g1[k][t * NV + v] * aw;
      }
#pragma unroll
      for (int k = 0; k < 4; ++k) {
        col[k]     = (_Float16)acc[k];
        col[4 + k] = (_Float16)sOv.pre.xc[k][n];
      }
    }
    *(v8h*)&sU[n][0] = col;
  }
  __syncthreads();   // sOv.pre dead; sOv.ht takes over

  // zero sHT K-padding (t = 60..63); never overwritten afterwards
  for (int i = tid; i < NOV; i += 256)
    *(v4h*)&sOv.ht[i][60] = (v4h){};

  // te1^T B-fragments (K=t in 0..63, N=j in 0..9; zero-padded), kept in registers
  ABu bte0, bte1;
  {
    int j = lh;
#pragma unroll
    for (int h = 0; h < 16; ++h) {
      int t0 = (h & 7) + (hi << 3) + ((h >> 3) << 4);
      int t1 = 32 + (h & 7) + (hi << 3) + ((h >> 3) << 4);
      bte0.v[h] = (_Float16)((j < 10) ? te1_w[j * NTC + t0] : 0.f);
      bte1.v[h] = (_Float16)((j < 10 && t1 < NTC) ? te1_w[j * NTC + t1] : 0.f);
    }
  }
  __syncthreads();

  const v8h zh = {};
  for (int ot = 0; ot < 16; ++ot) {
    // ---- phase 1: h = PReLU(W8 . u + C) via WMMA, K=8 in K=32 ----
    ABu afrag;
    {
      const _Float16* asrc = (hi == 0) ? &sW8[ot * 16 + lh][0] : &sZeroH[0];
      afrag.h[0] = *(const v8h*)asrc;
      afrag.h[1] = zh;
    }
    v4f cb0 = *(const v4f*)&sCb[ot * 16 + hi * 8];
    v4f cb1 = *(const v4f*)&sCb[ot * 16 + hi * 8 + 4];

    for (int nt = wave; nt < NTILES_N; nt += 16) {
      int n0 = nt * 16 + lh;
      int n1 = n0 + 128;
      ABu b0, b1;
      b0.h[0] = *(const v8h*)((hi == 0) ? &sU[n0][0] : &sZeroH[0]);
      b0.h[1] = zh;
      b1.h[0] = *(const v8h*)((hi == 0) ? &sU[n1][0] : &sZeroH[0]);
      b1.h[1] = zh;
      v8f c0 = {}, c1 = {};
      c0 = __builtin_amdgcn_wmma_f32_16x16x32_f16(false, afrag.v, false, b0.v,
                                                  (short)0, c0, false, false);
      c1 = __builtin_amdgcn_wmma_f32_16x16x32_f16(false, afrag.v, false, b1.v,
                                                  (short)0, c1, false, false);
      // PReLU + store in A-fragment-ready layout: row = o_local*21+v, col = t
      if (n0 < NTV) {
        int t = n0 / NV, v = n0 % NV;
#pragma unroll
        for (int r = 0; r < 4; ++r) {
          float hv = c0[r] + cb0[r];
          hv = (hv >= 0.f) ? hv : pa * hv;
          sOv.ht[(hi * 8 + r) * NV + v][t] = (_Float16)hv;
        }
#pragma unroll
        for (int r = 0; r < 4; ++r) {
          float hv = c0[4 + r] + cb1[r];
          hv = (hv >= 0.f) ? hv : pa * hv;
          sOv.ht[(hi * 8 + 4 + r) * NV + v][t] = (_Float16)hv;
        }
      }
      if (n1 < NTV) {
        int t = n1 / NV, v = n1 % NV;
#pragma unroll
        for (int r = 0; r < 4; ++r) {
          float hv = c1[r] + cb0[r];
          hv = (hv >= 0.f) ? hv : pa * hv;
          sOv.ht[(hi * 8 + r) * NV + v][t] = (_Float16)hv;
        }
#pragma unroll
        for (int r = 0; r < 4; ++r) {
          float hv = c1[4 + r] + cb1[r];
          hv = (hv >= 0.f) ? hv : pa * hv;
          sOv.ht[(hi * 8 + 4 + r) * NV + v][t] = (_Float16)hv;
        }
      }
    }
    __syncthreads();

    // ---- phase 2: te1 as WMMA: Z1[ov, j] = sum_t ht[ov,t] * te1_w[j,t] ----
    for (int m = wave; m < 21; m += 8) {
      const _Float16* arow = &sOv.ht[m * 16 + lh][hi * 8];
      ABu a0, a1;
      a0.h[0] = *(const v8h*)(arow);
      a0.h[1] = *(const v8h*)(arow + 16);
      a1.h[0] = *(const v8h*)(arow + 32);
      a1.h[1] = *(const v8h*)(arow + 48);
      v8f z = {};
      z = __builtin_amdgcn_wmma_f32_16x16x32_f16(false, a0.v, false, bte0.v,
                                                 (short)0, z, false, false);
      z = __builtin_amdgcn_wmma_f32_16x16x32_f16(false, a1.v, false, bte1.v,
                                                 (short)0, z, false, false);
      if (lh < 10) {
        float bj = sTeB[lh];
#pragma unroll
        for (int r = 0; r < 8; ++r)
          sZ1[m * 16 + hi * 8 + r][lh] = fast_tanh(z[r] + bj);
      }
    }
    __syncthreads();

    // ---- phase 3: z2[ov] = tanh(Z1 . te2_w + b) ----
    for (int p = tid; p < NOV; p += 256) {
      float acc = sTe2b;
#pragma unroll
      for (int j = 0; j < 10; ++j) acc += sZ1[p][j] * sTe2w[j];
      sZ2[p / NV][p % NV] = fast_tanh(acc);
    }
    __syncthreads();

    // ---- phase 4: node encoder -> hcont ----
    if (tid < 16) {
      float hc = sNe2b;
#pragma unroll
      for (int j = 0; j < 10; ++j) {
        float a = sNeB[j];
        for (int v = 0; v < NV; ++v) a += sZ2[tid][v] * sNe1w[j * NV + v];
        hc += fast_tanh(a) * sNe2w[j];
      }
      ws_hcont[b * NHD + ot * 16 + tid] = fast_tanh(hc);
    }
    __syncthreads();
  }
}

// ---------------------------------------------------------------------------
// Kernel 4: residual MLP head + IDCT.  16 blocks x 16 batch rows.
// ---------------------------------------------------------------------------
template <int IN_DIM, int OUT_DIM>
__device__ __forceinline__ void mlp_layer(const _Float16* __restrict__ actH,
                                          float* __restrict__ outF,
                                          _Float16* __restrict__ outH,
                                          const _Float16* __restrict__ Wp,
                                          const float* __restrict__ bias,
                                          bool do_tanh, const float* __restrict__ resF,
                                          int tid) {
  constexpr int KT = (IN_DIM + 31) / 32;
  constexpr int NT = (OUT_DIM + 15) / 16;
  constexpr bool FULL = (NT == 16);
  const int lane = tid & 31, wave = tid >> 5;
  const int lh = lane & 15, hi = (lane >> 4) & 1;
  const int nt0 = wave, nt1 = wave + 8;
  const bool has2 = FULL || (nt1 < NT);

  v8f c0 = {}, c1 = {};
  const _Float16* t0 = Wp + (size_t)(nt0 * KT) * 512 + lane * 16;
  const _Float16* t1 = Wp + (size_t)(nt1 * KT) * 512 + lane * 16;
  const _Float16* arow = actH + lh * 448 + hi * 8;
#pragma unroll
  for (int kt = 0; kt < KT; ++kt) {
    ABu a, b0;
    a.h[0]  = *(const v8h*)(arow + kt * 32);
    a.h[1]  = *(const v8h*)(arow + kt * 32 + 16);
    b0.h[0] = *(const v8h*)(t0 + kt * 512);
    b0.h[1] = *(const v8h*)(t0 + kt * 512 + 8);
    c0 = __builtin_amdgcn_wmma_f32_16x16x32_f16(false, a.v, false, b0.v,
                                                (short)0, c0, false, false);
    if (FULL || has2) {
      ABu b1;
      b1.h[0] = *(const v8h*)(t1 + kt * 512);
      b1.h[1] = *(const v8h*)(t1 + kt * 512 + 8);
      c1 = __builtin_amdgcn_wmma_f32_16x16x32_f16(false, a.v, false, b1.v,
                                                  (short)0, c1, false, false);
    }
  }

  auto epi = [&](const v8f& cf, int nt) {
    const int o = nt * 16 + lh;
#pragma unroll
    for (int r = 0; r < 8; ++r) {
      int row = hi * 8 + r;
      if (o < OUT_DIM) {
        float v = cf[r] + bias[o];
        if (do_tanh) v = fast_tanh(v);
        if (resF) v += resF[row * 448 + o];
        outF[row * 448 + o] = v;
        if (outH) outH[row * 448 + o] = (_Float16)v;
      }
    }
  };
  epi(c0, nt0);
  if (has2) epi(c1, nt1);
}

__global__ void __launch_bounds__(256)
k_head(const float* __restrict__ ws_dct, const float* __restrict__ ws_hcont,
       const _Float16* __restrict__ wp_g1, const float* __restrict__ g1_b,
       const _Float16* __restrict__ wp_g2a, const float* __restrict__ g2a_b,
       const _Float16* __restrict__ wp_g2b, const float* __restrict__ g2b_b,
       const _Float16* __restrict__ wp_g3a, const float* __restrict__ g3a_b,
       const _Float16* __restrict__ wp_g3b, const float* __restrict__ g3b_b,
       const _Float16* __restrict__ wp_g4, const float* __restrict__ g4_b,
       const float* __restrict__ idct_m, float* __restrict__ out) {
  __shared__ float    sA[16 * 448];
  __shared__ float    sB[16 * 448];
  __shared__ float    sC[16 * 448];
  __shared__ _Float16 sAh[16 * 448];
  __shared__ _Float16 sBh[16 * 448];
  __shared__ _Float16 sCh[16 * 448];
  const int bb = blockIdx.x * 16;
  const int tid = threadIdx.x;

  for (int i = tid; i < 16 * 448; i += 256) {
    int row = i / 448, col = i % 448;
    float v = 0.f;
    if (col < 180)      v = ws_dct[(bb + row) * 180 + col];
    else if (col < 436) v = ws_hcont[(bb + row) * NHD + (col - 180)];
    sA[i] = v;
    sAh[i] = (_Float16)v;
  }
  __syncthreads();

  mlp_layer<436, 256>(sAh, sB, sBh, wp_g1, g1_b, true, nullptr, tid);  // r1 = sB
  __syncthreads();
  mlp_layer<256, 256>(sBh, sC, sCh, wp_g2a, g2a_b, true, nullptr, tid);
  __syncthreads();
  mlp_layer<256, 256>(sCh, sA, sAh, wp_g2b, g2b_b, true, sB, tid);     // r2 = sA
  __syncthreads();
  mlp_layer<256, 256>(sAh, sB, sBh, wp_g3a, g3a_b, true, nullptr, tid);
  __syncthreads();
  mlp_layer<256, 256>(sBh, sC, sCh, wp_g3b, g3b_b, true, sA, tid);     // r3 = sC
  __syncthreads();
  mlp_layer<256, 180>(sCh, sA, (_Float16*)nullptr, wp_g4, g4_b, false, nullptr, tid);
  __syncthreads();

  for (int i = tid; i < 16 * 180; i += 256) {
    int row = i / 180, col = i % 180;
    sA[row * 448 + col] += ws_dct[(bb + row) * 180 + col];
  }
  __syncthreads();

  for (int i = tid; i < 16 * 90 * 3; i += 256) {
    int row = i / 270, r2 = i % 270, t = r2 / 3, k = r2 % 3;
    float acc = 0.f;
    for (int d = 0; d < NDCT; ++d)
      acc += idct_m[t * NDCT + d] * sA[row * 448 + d * 3 + k];
    out[(bb + row) * 270 + r2] = acc;
  }
}

// ---------------------------------------------------------------------------
extern "C" void kernel_launch(void* const* d_in, const int* in_sizes, int n_in,
                              void* d_out, int out_size, void* d_ws, size_t ws_size,
                              hipStream_t stream) {
  const float* x         = (const float*)d_in[0];
  const float* cont      = (const float*)d_in[1];
  const float* dct_m     = (const float*)d_in[3];
  const float* idct_m    = (const float*)d_in[4];
  const int*   root_idx  = (const int*)d_in[5];
  const float* T_c       = (const float*)d_in[7];
  const float* A_c       = (const float*)d_in[8];
  const float* conv_w    = (const float*)d_in[9];
  const float* conv_b    = (const float*)d_in[10];
  const float* bn1_scale = (const float*)d_in[11];
  const float* bn1_shift = (const float*)d_in[12];
  const float* resconv_w = (const float*)d_in[13];
  const float* resconv_b = (const float*)d_in[14];
  const float* bnr_scale = (const float*)d_in[15];
  const float* bnr_shift = (const float*)d_in[16];
  const float* prelu_a   = (const float*)d_in[17];
  const float* te1_w = (const float*)d_in[18];
  const float* te1_b = (const float*)d_in[19];
  const float* te2_w = (const float*)d_in[20];
  const float* te2_b = (const float*)d_in[21];
  const float* ne1_w = (const float*)d_in[22];
  const float* ne1_b = (const float*)d_in[23];
  const float* ne2_w = (const float*)d_in[24];
  const float* ne2_b = (const float*)d_in[25];
  const float* g1_w  = (const float*)d_in[26];
  const float* g1_b  = (const float*)d_in[27];
  const float* g2a_w = (const float*)d_in[28];
  const float* g2a_b = (const float*)d_in[29];
  const float* g2b_w = (const float*)d_in[30];
  const float* g2b_b = (const float*)d_in[31];
  const float* g3a_w = (const float*)d_in[32];
  const float* g3a_b = (const float*)d_in[33];
  const float* g3b_w = (const float*)d_in[34];
  const float* g3b_b = (const float*)d_in[35];
  const float* g4_w  = (const float*)d_in[36];
  const float* g4_b  = (const float*)d_in[37];

  float* ws_dct   = (float*)d_ws;          // 256*180 f32
  float* ws_hcont = ws_dct + NBS * 180;    // 256*256 f32
  _Float16* wp    = (_Float16*)(ws_hcont + NBS * NHD);  // packed head weights

  const int L_G1  = 16 * 14 * 512;   // 436->256
  const int L_G2  = 16 * 8  * 512;   // 256->256
  const int L_G4  = 12 * 8  * 512;   // 256->180
  _Float16* wp_g1  = wp;
  _Float16* wp_g2a = wp_g1  + L_G1;
  _Float16* wp_g2b = wp_g2a + L_G2;
  _Float16* wp_g3a = wp_g2b + L_G2;
  _Float16* wp_g3b = wp_g3a + L_G2;
  _Float16* wp_g4  = wp_g3b + L_G2;

  k_pack<<<(L_G1 + 255) / 256, 256, 0, stream>>>(g1_w,  wp_g1,  436, 256);
  k_pack<<<(L_G2 + 255) / 256, 256, 0, stream>>>(g2a_w, wp_g2a, 256, 256);
  k_pack<<<(L_G2 + 255) / 256, 256, 0, stream>>>(g2b_w, wp_g2b, 256, 256);
  k_pack<<<(L_G2 + 255) / 256, 256, 0, stream>>>(g3a_w, wp_g3a, 256, 256);
  k_pack<<<(L_G2 + 255) / 256, 256, 0, stream>>>(g3b_w, wp_g3b, 256, 256);
  k_pack<<<(L_G4 + 255) / 256, 256, 0, stream>>>(g4_w,  wp_g4,  256, 180);

  k_dct<<<NBS, 192, 0, stream>>>(x, dct_m, root_idx, ws_dct);
  k_gcn<<<NBS, 256, 0, stream>>>(cont, T_c, A_c, conv_w, conv_b, bn1_scale, bn1_shift,
                                 resconv_w, resconv_b, bnr_scale, bnr_shift, prelu_a,
                                 te1_w, te1_b, te2_w, te2_b,
                                 ne1_w, ne1_b, ne2_w, ne2_b, ws_hcont);
  k_head<<<16, 256, 0, stream>>>(ws_dct, ws_hcont,
                                 wp_g1, g1_b, wp_g2a, g2a_b, wp_g2b, g2b_b,
                                 wp_g3a, g3a_b, wp_g3b, g3b_b, wp_g4, g4_b,
                                 idct_m, (float*)d_out);
}
